// LogicConv2d_21208548507913
// MI455X (gfx1250) — matrix operand order repaired and checked by reading, not compiled, not run on hardware
//
#include <hip/hip_runtime.h>

typedef _Float16 v8h  __attribute__((ext_vector_type(8)));
typedef _Float16 v16h __attribute__((ext_vector_type(16)));
typedef float    v8f  __attribute__((ext_vector_type(8)));

#define CIN   64
#define COUT  128
#define HW    56
#define DTOT  576          // CIN * 9, K dimension
#define LTOT  3136         // 56*56
#define PIXS  72           // channel-last pixel stride in f16 units (144B, 16B aligned)
#define TROWS 4            // input rows staged per block (2 output rows + halo)
#define NPIX  (TROWS * HW) // real pixels in tile
#define ZPIX  NPIX         // extra zero-filled pixel slot for padding taps
// LDS per buffer: (NPIX+1)*PIXS f16 = 16200 halves = 32400 B; hi+lo = 64800 B < 64 KB

// ---------------------------------------------------------------------------
// Prep: reorder W from d = c*9 + khw  ->  d' = khw*64 + c, split into f16 hi/lo
// ---------------------------------------------------------------------------
__global__ void wprep_f16split(const float* __restrict__ W,
                               _Float16* __restrict__ whi,
                               _Float16* __restrict__ wlo) {
  int i = blockIdx.x * 256 + threadIdx.x;
  if (i >= COUT * DTOT) return;
  int o   = i / DTOT;
  int dp  = i - o * DTOT;       // d' = khw*64 + c
  int khw = dp >> 6;
  int c   = dp & 63;
  int d   = c * 9 + khw;        // original flat index (C, kh, kw ordering)
  float w = W[o * DTOT + d];
  _Float16 hi = (_Float16)w;
  _Float16 lo = (_Float16)(w - (float)hi);
  whi[i] = hi;
  wlo[i] = lo;
}

// ---------------------------------------------------------------------------
// Main: one block = (batch b, output-row pair oh0..oh0+1); 4 waves.
// Wave w: output row oh0 + (w>>1), columns (w&1)*32 .. +31 (2 N-tiles of 16).
// Each wave computes all 128 out-channels via split-f16 WMMA (f32 accum).
// ---------------------------------------------------------------------------
__launch_bounds__(128)
__global__ void logic_conv_wmma(const float* __restrict__ x,
                                const _Float16* __restrict__ whi,
                                const _Float16* __restrict__ wlo,
                                const float* __restrict__ bias,
                                float* __restrict__ out) {
  __shared__ _Float16 sh_hi[(NPIX + 1) * PIXS];
  __shared__ _Float16 sh_lo[(NPIX + 1) * PIXS];

  const int oh0 = blockIdx.x * 2;  // first output row of the pair (0..54)
  const int b   = blockIdx.y;      // 0..31
  const int t   = threadIdx.x;     // 0..127

  // ---- stage 4 input rows (oh0-1 .. oh0+2), channel-last, f16 hi/lo ----
  // No pad columns: out-of-range rows stored as zero; column pad uses ZPIX.
  for (int e = t; e < TROWS * HW * CIN; e += 128) {
    int iw  = e % HW;
    int idx = e / HW;
    int c   = idx & 63;
    int r   = idx >> 6;            // tile row 0..3
    int ihg = oh0 - 1 + r;
    float v = 0.0f;
    if ((unsigned)ihg < 56u)
      v = x[(((b * CIN + c) * HW) + ihg) * HW + iw];
    _Float16 hi = (_Float16)v;
    _Float16 lo = (_Float16)(v - (float)hi);
    int pix = r * HW + iw;
    sh_hi[pix * PIXS + c] = hi;
    sh_lo[pix * PIXS + c] = lo;
  }
  if (t < CIN) {                   // zero pixel for padding taps
    sh_hi[ZPIX * PIXS + t] = (_Float16)0.0f;
    sh_lo[ZPIX * PIXS + t] = (_Float16)0.0f;
  }
  __syncthreads();

  const int wave = t >> 5;
  const int lane = t & 31;
  const int col  = lane & 15;      // N index within a 16x16 tile
  const int ksel = lane >> 4;      // which K half this lane holds
  const int wr   = wave >> 1;      // which output row of the pair
  const int cb   = (wave & 1) * 32;// column base for this wave
  const int oh   = oh0 + wr;
  const int arow = lane & 15;      // M row this lane holds in A fragments

  // ---- 16 accumulator tiles (2 N x 8 M), bias folded in ----
  v8f acc[16];
  #pragma unroll
  for (int n = 0; n < 2; ++n) {
    #pragma unroll
    for (int m = 0; m < 8; ++m) {
      #pragma unroll
      for (int r = 0; r < 8; ++r)
        acc[n * 8 + m][r] = bias[m * 16 + r + ksel * 8];
    }
  }

  // ---- K loop: 18 steps of 32 over d' = khw*64 + c ----
  #pragma unroll 2
  for (int step = 0; step < 18; ++step) {
    const int kk  = step * 32;
    const int khw = kk >> 6;            // tap index for this step
    const int c0  = (step & 1) << 5;    // channel base: 0 or 32
    const int kh  = khw / 3;
    const int kw  = khw - kh * 3;

    // B fragments for the 2 N-tiles (lanes 0-15: K={0..7,16..23}, lanes 16-31: K={8..15,24..31})
    v16h bhi[2], blo[2];
    #pragma unroll
    for (int n = 0; n < 2; ++n) {
      const int ow  = cb + n * 16 + col;
      const int iwg = ow + kw - 1;                       // input column of this tap
      const int pix = ((unsigned)iwg < 56u && ow < 56)
                        ? ((wr + kh) * HW + iwg)         // in-tile pixel
                        : ZPIX;                          // zero pixel (pad / masked)
      const int boff = pix * PIXS + c0 + ksel * 8;
      *(v8h*)&bhi[n]         = *(const v8h*)(sh_hi + boff);
      *(((v8h*)&bhi[n]) + 1) = *(const v8h*)(sh_hi + boff + 16);
      *(v8h*)&blo[n]         = *(const v8h*)(sh_lo + boff);
      *(((v8h*)&blo[n]) + 1) = *(const v8h*)(sh_lo + boff + 16);
    }

    #pragma unroll
    for (int m = 0; m < 8; ++m) {
      // A fragments: lanes 0-15 K={0..7,16..23}, lanes 16-31 K={8..15,24..31}
      const int o    = m * 16 + arow;
      const int aoff = o * DTOT + kk + ksel * 8;
      v16h ahi, alo;
      *(v8h*)&ahi         = *(const v8h*)(whi + aoff);
      *(((v8h*)&ahi) + 1) = *(const v8h*)(whi + aoff + 16);
      *(v8h*)&alo         = *(const v8h*)(wlo + aoff);
      *(((v8h*)&alo) + 1) = *(const v8h*)(wlo + aoff + 16);

      #pragma unroll
      for (int n = 0; n < 2; ++n) {
        // split-f16 product: hi*hi + hi*lo + lo*hi, f32 accumulate
        acc[n * 8 + m] = __builtin_amdgcn_wmma_f32_16x16x32_f16(
            false, ahi, false, bhi[n], (short)0, acc[n * 8 + m], false, false);
        acc[n * 8 + m] = __builtin_amdgcn_wmma_f32_16x16x32_f16(
            false, ahi, false, blo[n], (short)0, acc[n * 8 + m], false, false);
        acc[n * 8 + m] = __builtin_amdgcn_wmma_f32_16x16x32_f16(
            false, alo, false, bhi[n], (short)0, acc[n * 8 + m], false, false);
      }
    }
  }

  // ---- epilogue: intensity = sin^2(z); hard binarize at 0.5; coalesced store ----
  #pragma unroll
  for (int n = 0; n < 2; ++n) {
    const int ow = cb + n * 16 + col;
    if (ow < 56) {
      const int lbase = oh * HW + ow;
      #pragma unroll
      for (int m = 0; m < 8; ++m) {
        #pragma unroll
        for (int r = 0; r < 8; ++r) {
          int o = m * 16 + r + ksel * 8;
          float s = __sinf(acc[n * 8 + m][r]);
          float inten = s * s;
          out[(b * COUT + o) * LTOT + lbase] = inten > 0.5f ? 1.0f : 0.0f;
        }
      }
    }
  }
}

// ---------------------------------------------------------------------------
extern "C" void kernel_launch(void* const* d_in, const int* in_sizes, int n_in,
                              void* d_out, int out_size, void* d_ws, size_t ws_size,
                              hipStream_t stream) {
  const float* x    = (const float*)d_in[0];   // (32, 64, 56, 56)
  const float* W    = (const float*)d_in[1];   // (128, 576)
  const float* bias = (const float*)d_in[2];   // (128,)
  float* out = (float*)d_out;                  // (32, 128, 56, 56)

  _Float16* whi = (_Float16*)d_ws;             // 128*576 f16
  _Float16* wlo = whi + COUT * DTOT;           // 128*576 f16  (ws needs ~288 KB)

  wprep_f16split<<<(COUT * DTOT + 255) / 256, 256, 0, stream>>>(W, whi, wlo);

  dim3 grid(28, 32);   // (output-row pair, batch)
  logic_conv_wmma<<<grid, 128, 0, stream>>>(x, whi, wlo, bias, out);
}
